// NaViTEncoder_89799176225228
// MI455X (gfx1250) — compile-verified, matrix operand order checked by reading, not run on hardware
//
#include <hip/hip_runtime.h>
#include <hip/hip_bf16.h>
#include <cstdint>

// ---------------------------------------------------------------------------
// NaViT encoder for gfx1250 (MI455X): bf16 WMMA GEMMs + flash attention.
// ---------------------------------------------------------------------------

typedef __bf16 bf16;
typedef __attribute__((ext_vector_type(16))) __bf16 bf16x16;
typedef __attribute__((ext_vector_type(8)))  __bf16 bf16x8;
typedef __attribute__((ext_vector_type(4)))  __bf16 bf16x4;
typedef __attribute__((ext_vector_type(8)))  float  f32x8;

#define WMMA_BF16(a, b, c) \
  __builtin_amdgcn_wmma_f32_16x16x32_bf16(false, (a), false, (b), (short)0, (c), false, false)

static constexpr int Bn  = 8;
static constexpr int Sn  = 1024;
static constexpr int Dn  = 768;
static constexpr int Hn  = 12;
static constexpr int Ln  = 12;
static constexpr int HDn = 64;
static constexpr int Fn  = 3072;
static constexpr int Mtok = Bn * Sn;   // 8192 tokens

// ---------------------------------------------------------------------------
// fp32 -> bf16 conversion, 4 elements / thread (n is a multiple of 4)
// ---------------------------------------------------------------------------
__global__ void f32_to_bf16_kernel(const float* __restrict__ in,
                                   bf16* __restrict__ out, long n4) {
  long i = (long)blockIdx.x * blockDim.x + threadIdx.x;
  if (i < n4) {
    const float4 v = ((const float4*)in)[i];
    bf16x4 o;
    o[0] = (bf16)v.x; o[1] = (bf16)v.y; o[2] = (bf16)v.z; o[3] = (bf16)v.w;
    ((bf16x4*)out)[i] = o;
  }
}

// ---------------------------------------------------------------------------
// Segment ids from block-diagonal mask: segq[q] = #boundaries in mask
// sub-diagonal up to q. One 1024-thread block, Hillis-Steele scan.
// ---------------------------------------------------------------------------
__global__ void __launch_bounds__(1024)
seg_kernel(const unsigned char* __restrict__ mask, int* __restrict__ segq) {
  __shared__ int sh[Sn];
  const int t = threadIdx.x;
  sh[t] = (t > 0 && mask[(long)t * Sn + (t - 1)]) ? 1 : 0;
  __syncthreads();
  for (int ofs = 1; ofs < Sn; ofs <<= 1) {
    int v = (t >= ofs) ? sh[t - ofs] : 0;
    __syncthreads();
    sh[t] += v;
    __syncthreads();
  }
  segq[t] = sh[t];
}

// ---------------------------------------------------------------------------
// LayerNorm over D=768: one wave per token, 24 values/lane, shfl reductions.
// OUT_F32=false -> bf16 normed activations; true -> fp32 (final LN to d_out).
// ---------------------------------------------------------------------------
template <bool OUT_F32>
__global__ void __launch_bounds__(256)
ln_kernel(const float* __restrict__ x, const float* __restrict__ w,
          const float* __restrict__ b, void* __restrict__ out) {
  const int tok  = blockIdx.x * 8 + (threadIdx.x >> 5);
  const int lane = threadIdx.x & 31;
  const float* row = x + (long)tok * Dn;

  float vals[24];
  float s = 0.f;
#pragma unroll
  for (int i = 0; i < 24; ++i) { vals[i] = row[lane + i * 32]; s += vals[i]; }
#pragma unroll
  for (int m = 16; m; m >>= 1) s += __shfl_xor(s, m);
  const float mean = s * (1.0f / Dn);

  float var = 0.f;
#pragma unroll
  for (int i = 0; i < 24; ++i) { float d = vals[i] - mean; var += d * d; }
#pragma unroll
  for (int m = 16; m; m >>= 1) var += __shfl_xor(var, m);
  const float rstd = rsqrtf(var * (1.0f / Dn) + 1e-5f);

#pragma unroll
  for (int i = 0; i < 24; ++i) {
    int c = lane + i * 32;
    float y = (vals[i] - mean) * rstd * w[c] + b[c];
    if (OUT_F32)
      ((float*)out)[(long)tok * Dn + c] = y;
    else
      ((bf16*)out)[(long)tok * Dn + c] = (bf16)y;
  }
}

// ---------------------------------------------------------------------------
// GEMM: C = epi(A[M,K]bf16 @ W[K,N]bf16 + bias)
// Block tile 128x128, 8 waves in 4x2 grid, wave tile 32x64 (2x4 WMMA frags).
// EPI: 0 = bf16 store, 1 = exact-GELU -> bf16, 2 = residual + acc -> fp32
// ---------------------------------------------------------------------------
template <int EPI>
__global__ void __launch_bounds__(256)
gemm_kernel(const bf16* __restrict__ A, const bf16* __restrict__ W,
            const float* __restrict__ bias, void* __restrict__ Cout,
            const float* __restrict__ residual, int M, int N, int K) {
  constexpr int LDA = 40;                // padded bf16 row stride (80B, 16B-aligned)
  __shared__ bf16 As[128 * LDA];         // A tile 128x32
  __shared__ bf16 Bt[128 * LDA];         // B tile transposed: [n][k], 128x32

  const int tid   = threadIdx.x;
  const int lane  = tid & 31;
  const int wave  = tid >> 5;
  const int wrow  = wave >> 1;           // 0..3 (32-row strips)
  const int wcol  = wave & 1;            // 0..1 (64-col strips)
  const int m0    = blockIdx.y * 128;
  const int n0    = blockIdx.x * 128;
  const int lhalf = lane >> 4;           // 0/1 : selects K sub-pattern
  const int l15   = lane & 15;

  f32x8 acc[2][4] = {};

  for (int k0 = 0; k0 < K; k0 += 32) {
    // ---- stage A tile: 128x32, 512 chunks of 8 bf16 (2 b128 per thread)
#pragma unroll
    for (int it = 0; it < 2; ++it) {
      int c   = tid + it * 256;
      int row = c >> 2;
      int cc  = (c & 3) * 8;
      const bf16* src = A + (long)(m0 + row) * K + k0 + cc;
      if (k0 + 32 < K) __builtin_prefetch(src + 32, 0, 1);
      *(bf16x8*)&As[row * LDA + cc] = *(const bf16x8*)src;
    }
    // ---- stage B tile 32x128, transposed into Bt[n][k]
#pragma unroll
    for (int it = 0; it < 2; ++it) {
      int c  = tid + it * 256;           // 512 chunks of 8
      int kr = c >> 4;                   // 0..31
      int nc = (c & 15) * 8;             // 0..120
      const bf16* src = W + (long)(k0 + kr) * N + n0 + nc;
      if (k0 + 32 < K) __builtin_prefetch(src + 32 * N, 0, 1);
      bf16x8 vv = *(const bf16x8*)src;
#pragma unroll
      for (int j = 0; j < 8; ++j) Bt[(nc + j) * LDA + kr] = vv[j];
    }
    __syncthreads();

    // ---- fragments (ISA 16-bit A layout: lane=row, half-wave K split)
    bf16x16 af[2], bfrag[4];
#pragma unroll
    for (int mi = 0; mi < 2; ++mi) {
      int m = wrow * 32 + mi * 16 + l15;
      bf16x8 lo = *(const bf16x8*)&As[m * LDA + lhalf * 8];
      bf16x8 hi = *(const bf16x8*)&As[m * LDA + lhalf * 8 + 16];
#pragma unroll
      for (int j = 0; j < 8; ++j) { af[mi][j] = lo[j]; af[mi][j + 8] = hi[j]; }
    }
#pragma unroll
    for (int ni = 0; ni < 4; ++ni) {
      int n = wcol * 64 + ni * 16 + l15;
      bf16x8 lo = *(const bf16x8*)&Bt[n * LDA + lhalf * 8];
      bf16x8 hi = *(const bf16x8*)&Bt[n * LDA + lhalf * 8 + 16];
#pragma unroll
      for (int j = 0; j < 8; ++j) { bfrag[ni][j] = lo[j]; bfrag[ni][j + 8] = hi[j]; }
    }
#pragma unroll
    for (int mi = 0; mi < 2; ++mi)
#pragma unroll
      for (int ni = 0; ni < 4; ++ni)
        acc[mi][ni] = WMMA_BF16(af[mi], bfrag[ni], acc[mi][ni]);
    __syncthreads();
  }

  // ---- epilogue: C layout lane=col, row = r + 8*(lane>>4)
#pragma unroll
  for (int mi = 0; mi < 2; ++mi) {
#pragma unroll
    for (int ni = 0; ni < 4; ++ni) {
      int col = n0 + wcol * 64 + ni * 16 + l15;
      float bv = bias ? bias[col] : 0.f;
#pragma unroll
      for (int r = 0; r < 8; ++r) {
        int row = m0 + wrow * 32 + mi * 16 + r + 8 * lhalf;
        float v = acc[mi][ni][r] + bv;
        long idx = (long)row * N + col;
        if (EPI == 0) {
          ((bf16*)Cout)[idx] = (bf16)v;
        } else if (EPI == 1) {
          ((bf16*)Cout)[idx] = (bf16)(0.5f * v * (1.0f + erff(v * 0.70710678118f)));
        } else {
          ((float*)Cout)[idx] = residual[idx] + v;
        }
      }
    }
  }
}

// ---------------------------------------------------------------------------
// Flash attention: one wave per (b, h, 16-query tile). Key blocks of 32.
// qkv bf16 [B,S,3D]; segq int [S]; out bf16 [B,S,D].
// ---------------------------------------------------------------------------
__global__ void __launch_bounds__(32)
attn_kernel(const bf16* __restrict__ qkv, const int* __restrict__ segq,
            bf16* __restrict__ out) {
  const int q0 = blockIdx.x * 16;
  const int h  = blockIdx.y;
  const int b  = blockIdx.z;
  const int lane  = threadIdx.x;
  const int lhalf = lane >> 4;
  const int l15   = lane & 15;

  __shared__ bf16 Pl[16 * 32];           // P tile (16 q x 32 k), for A-layout reload
  __shared__ bf16 Vt[64 * 32];           // V^T tile (64 dims x 32 keys)

  const long rs = 3 * Dn;                // qkv row stride
  const bf16* base = qkv + (long)b * Sn * rs;

  // Q fragments (16x64 -> two 16x32 A fragments), kept in registers
  bf16x16 qa[2];
  {
    const bf16* qp = base + (long)(q0 + l15) * rs + h * HDn;
#pragma unroll
    for (int f = 0; f < 2; ++f) {
      bf16x8 lo = *(const bf16x8*)(qp + f * 32 + lhalf * 8);
      bf16x8 hi = *(const bf16x8*)(qp + f * 32 + lhalf * 8 + 16);
#pragma unroll
      for (int j = 0; j < 8; ++j) { qa[f][j] = lo[j]; qa[f][j + 8] = hi[j]; }
    }
  }

  // segment id of the 8 rows this lane's C-layout slots cover (fixed all loop)
  int rseg[8];
#pragma unroll
  for (int r = 0; r < 8; ++r) rseg[r] = segq[q0 + r + 8 * lhalf];

  float mrow[8], srow[8];
#pragma unroll
  for (int r = 0; r < 8; ++r) { mrow[r] = -3.0e38f; srow[r] = 0.f; }
  f32x8 o[4] = {};

  for (int kb = 0; kb < Sn; kb += 32) {
    // ---- scores = Q @ K^T for two 16-key tiles
    f32x8 sc[2] = {};
#pragma unroll
    for (int t = 0; t < 2; ++t) {
      const bf16* kp = base + (long)(kb + t * 16 + l15) * rs + Dn + h * HDn;
#pragma unroll
      for (int f = 0; f < 2; ++f) {
        bf16x16 kf;
        bf16x8 lo = *(const bf16x8*)(kp + f * 32 + lhalf * 8);
        bf16x8 hi = *(const bf16x8*)(kp + f * 32 + lhalf * 8 + 16);
#pragma unroll
        for (int j = 0; j < 8; ++j) { kf[j] = lo[j]; kf[j + 8] = hi[j]; }
        sc[t] = WMMA_BF16(qa[f], kf, sc[t]);
      }
    }

    // segment id of this lane's key column in each 16-key tile
    const int kseg0 = segq[kb + l15];
    const int kseg1 = segq[kb + 16 + l15];

    // ---- mask + scale + online softmax (rows live in 16-lane halves)
    float pv[2][8];
    bool  pm[2][8];
#pragma unroll
    for (int r = 0; r < 8; ++r) {
      pm[0][r] = (rseg[r] != kseg0);
      pm[1][r] = (rseg[r] != kseg1);
#pragma unroll
      for (int t = 0; t < 2; ++t) {
        float v = sc[t][r] * 0.125f;     // 1/sqrt(64)
        pv[t][r] = pm[t][r] ? -3.0e38f : v;
      }
      float mx = fmaxf(pv[0][r], pv[1][r]);
      mx = fmaxf(mx, __shfl_xor(mx, 1));
      mx = fmaxf(mx, __shfl_xor(mx, 2));
      mx = fmaxf(mx, __shfl_xor(mx, 4));
      mx = fmaxf(mx, __shfl_xor(mx, 8));
      float newm  = fmaxf(mrow[r], mx);
      float alpha = __expf(mrow[r] - newm);
      // masked entries contribute exactly zero (avoids exp(-inf - -inf) = 1)
      float p0 = pm[0][r] ? 0.f : __expf(pv[0][r] - newm);
      float p1 = pm[1][r] ? 0.f : __expf(pv[1][r] - newm);
      pv[0][r] = p0; pv[1][r] = p1;
      float rsum = p0 + p1;
      rsum += __shfl_xor(rsum, 1);
      rsum += __shfl_xor(rsum, 2);
      rsum += __shfl_xor(rsum, 4);
      rsum += __shfl_xor(rsum, 8);
      srow[r] = srow[r] * alpha + rsum;
      mrow[r] = newm;
#pragma unroll
      for (int t2 = 0; t2 < 4; ++t2) o[t2][r] *= alpha;
    }

    // ---- P (C-layout) -> LDS, and V^T -> LDS
#pragma unroll
    for (int r = 0; r < 8; ++r) {
      int prow = r + 8 * lhalf;
      Pl[prow * 32 + l15]      = (bf16)pv[0][r];
      Pl[prow * 32 + 16 + l15] = (bf16)pv[1][r];
    }
    {
      const bf16* vp = base + (long)(kb + lane) * rs + 2 * Dn + h * HDn;
#pragma unroll
      for (int dc = 0; dc < 8; ++dc) {
        bf16x8 vv = *(const bf16x8*)(vp + dc * 8);
#pragma unroll
        for (int j = 0; j < 8; ++j) Vt[(dc * 8 + j) * 32 + lane] = vv[j];
      }
    }
    __syncthreads();

    // ---- O += P @ V (K=32 keys, four 16-dim N tiles)
    bf16x16 pa;
    {
      bf16x8 lo = *(const bf16x8*)&Pl[l15 * 32 + lhalf * 8];
      bf16x8 hi = *(const bf16x8*)&Pl[l15 * 32 + lhalf * 8 + 16];
#pragma unroll
      for (int j = 0; j < 8; ++j) { pa[j] = lo[j]; pa[j + 8] = hi[j]; }
    }
#pragma unroll
    for (int t2 = 0; t2 < 4; ++t2) {
      bf16x16 vb;
      int d = t2 * 16 + l15;
      bf16x8 lo = *(const bf16x8*)&Vt[d * 32 + lhalf * 8];
      bf16x8 hi = *(const bf16x8*)&Vt[d * 32 + lhalf * 8 + 16];
#pragma unroll
      for (int j = 0; j < 8; ++j) { vb[j] = lo[j]; vb[j + 8] = hi[j]; }
      o[t2] = WMMA_BF16(pa, vb, o[t2]);
    }
    __syncthreads();
  }

  // ---- normalize and store bf16
#pragma unroll
  for (int r = 0; r < 8; ++r) {
    int qrow = q0 + r + 8 * lhalf;
    float inv = 1.0f / srow[r];
#pragma unroll
    for (int t2 = 0; t2 < 4; ++t2) {
      out[((long)b * Sn + qrow) * Dn + h * HDn + t2 * 16 + l15] =
          (bf16)(o[t2][r] * inv);
    }
  }
}

// ---------------------------------------------------------------------------
// Host orchestration
// ---------------------------------------------------------------------------
static inline size_t align_up(size_t x) { return (x + 255) & ~(size_t)255; }

extern "C" void kernel_launch(void* const* d_in, const int* in_sizes, int n_in,
                              void* d_out, int out_size, void* d_ws, size_t ws_size,
                              hipStream_t stream) {
  const float* tok  = (const float*)d_in[0];
  const unsigned char* mask = (const unsigned char*)d_in[1];
  const float* ln1w = (const float*)d_in[2];
  const float* ln1b = (const float*)d_in[3];
  const float* wqkv = (const float*)d_in[4];
  const float* bqkv = (const float*)d_in[5];
  const float* wo   = (const float*)d_in[6];
  const float* bo   = (const float*)d_in[7];
  const float* ln2w = (const float*)d_in[8];
  const float* ln2b = (const float*)d_in[9];
  const float* w1   = (const float*)d_in[10];
  const float* b1   = (const float*)d_in[11];
  const float* w2   = (const float*)d_in[12];
  const float* b2   = (const float*)d_in[13];
  const float* nw   = (const float*)d_in[14];
  const float* nb   = (const float*)d_in[15];

  char* ws = (char*)d_ws;
  size_t off = 0;
  auto take = [&](size_t bytes) -> void* {
    void* p = ws + off;
    off = align_up(off + bytes);
    return p;
  };
  bf16*  wqkv_bf = (bf16*)take((size_t)Ln * Dn * 3 * Dn * sizeof(bf16));
  bf16*  wo_bf   = (bf16*)take((size_t)Ln * Dn * Dn * sizeof(bf16));
  bf16*  w1_bf   = (bf16*)take((size_t)Ln * Dn * Fn * sizeof(bf16));
  bf16*  w2_bf   = (bf16*)take((size_t)Ln * Fn * Dn * sizeof(bf16));
  float* x       = (float*)take((size_t)Mtok * Dn * sizeof(float));
  bf16*  nrm     = (bf16*)take((size_t)Mtok * Dn * sizeof(bf16));
  bf16*  qkvb    = (bf16*)take((size_t)Mtok * 3 * Dn * sizeof(bf16));
  bf16*  attnb   = (bf16*)take((size_t)Mtok * Dn * sizeof(bf16));
  bf16*  hb      = (bf16*)take((size_t)Mtok * Fn * sizeof(bf16));
  int*   segq    = (int*)take((size_t)Sn * sizeof(int));

  auto conv = [&](const float* src, bf16* dst, long n) {
    long n4 = n / 4;
    unsigned blocks = (unsigned)((n4 + 255) / 256);
    f32_to_bf16_kernel<<<dim3(blocks), dim3(256), 0, stream>>>(src, dst, n4);
  };
  conv(wqkv, wqkv_bf, (long)Ln * Dn * 3 * Dn);
  conv(wo,   wo_bf,   (long)Ln * Dn * Dn);
  conv(w1,   w1_bf,   (long)Ln * Dn * Fn);
  conv(w2,   w2_bf,   (long)Ln * Fn * Dn);

  seg_kernel<<<dim3(1), dim3(1024), 0, stream>>>(mask, segq);

  hipMemcpyAsync(x, tok, (size_t)Mtok * Dn * sizeof(float),
                 hipMemcpyDeviceToDevice, stream);

  const dim3 blk256(256), blk32(32);
  for (int l = 0; l < Ln; ++l) {
    // x -> ln1 -> nrm (bf16)
    ln_kernel<false><<<dim3(Mtok / 8), blk256, 0, stream>>>(
        x, ln1w + (long)l * Dn, ln1b + (long)l * Dn, nrm);
    // qkv = nrm @ Wqkv + bqkv  (bf16 out)
    gemm_kernel<0><<<dim3(3 * Dn / 128, Mtok / 128), blk256, 0, stream>>>(
        nrm, wqkv_bf + (long)l * Dn * 3 * Dn, bqkv + (long)l * 3 * Dn,
        qkvb, nullptr, Mtok, 3 * Dn, Dn);
    // attention
    attn_kernel<<<dim3(Sn / 16, Hn, Bn), blk32, 0, stream>>>(qkvb, segq, attnb);
    // x = x + attnb @ Wo + bo  (fp32)
    gemm_kernel<2><<<dim3(Dn / 128, Mtok / 128), blk256, 0, stream>>>(
        attnb, wo_bf + (long)l * Dn * Dn, bo + (long)l * Dn,
        x, x, Mtok, Dn, Dn);
    // x -> ln2 -> nrm
    ln_kernel<false><<<dim3(Mtok / 8), blk256, 0, stream>>>(
        x, ln2w + (long)l * Dn, ln2b + (long)l * Dn, nrm);
    // h = gelu(nrm @ W1 + b1)  (bf16)
    gemm_kernel<1><<<dim3(Fn / 128, Mtok / 128), blk256, 0, stream>>>(
        nrm, w1_bf + (long)l * Dn * Fn, b1 + (long)l * Fn,
        hb, nullptr, Mtok, Fn, Dn);
    // x = x + h @ W2 + b2  (fp32)
    gemm_kernel<2><<<dim3(Dn / 128, Mtok / 128), blk256, 0, stream>>>(
        hb, w2_bf + (long)l * Fn * Dn, b2 + (long)l * Dn,
        x, x, Mtok, Dn, Fn);
  }
  // final layernorm -> d_out (fp32)
  ln_kernel<true><<<dim3(Mtok / 8), blk256, 0, stream>>>(x, nw, nb, d_out);
}
